// Attention_10849087390107
// MI455X (gfx1250) — matrix operand order repaired and checked by reading, not compiled, not run on hardware
//
#include <hip/hip_runtime.h>
#include <hip/hip_bf16.h>

// ---------------------------------------------------------------------------
// MI455X (gfx1250) fused attention:
//   out = proj( softmax( mask(QK^T/8) ) V ),  Q,K,V = x @ w_qkv^T
// B=4, N=2048, DIM=1024, H=16, hd=64.  ~137 GFLOP vs ~100 MB minimal traffic
// -> compute bound -> all matmuls via v_wmma_f32_16x16x32_f16.
//   * GEMMs: 64x64 tile/wave (4x4 frags) = 16 WMMA per 8 fragment loads
//     (~32 FLOP/byte from L2; qkv operands are L2-resident at 192MB).
//   * V is scattered TRANSPOSED [b,h,d,n] by the QKV GEMM so flash attention's
//     P@V B-fragments are contiguous b128 global loads (no LDS transpose).
//   * Flash: 32 query rows/wave, K/V fragments shared by both 16-row subtiles;
//     scores never touch HBM; LDS only for P C-layout -> A-layout restage.
// ---------------------------------------------------------------------------

typedef _Float16 v8h  __attribute__((ext_vector_type(8)));
typedef _Float16 v16h __attribute__((ext_vector_type(16)));
typedef float    v8f  __attribute__((ext_vector_type(8)));

#define BATCH 4
#define SEQ   2048
#define DIMC  1024
#define HEADS 16
#define HD    64
#define QKVC  3072
#define NEG_INF (-3.402823466e38f)

__device__ __forceinline__ v8f vzero8f() {
  v8f v = {0.f,0.f,0.f,0.f,0.f,0.f,0.f,0.f};
  return v;
}

// A-fragment (16x32 f16): lane holds row M=lane&15; half0: K {0..7,16..23},
// half1: K {8..15,24..31}. Caller passes p = row_base + k0 + half*8.
__device__ __forceinline__ v16h load_a_frag(const _Float16* p) {
  v8h lo = *(const v8h*)(p);
  v8h hi = *(const v8h*)(p + 16);
  return __builtin_shufflevector(lo, hi, 0,1,2,3,4,5,6,7,8,9,10,11,12,13,14,15);
}

// B-fragment (32x16 f16): lane holds col N=lane&15; half0: K {0..7,16..23},
// half1: K {8..15,24..31}. Caller passes p = col_base + k0 + half*8.
__device__ __forceinline__ v16h load_b_frag(const _Float16* p) {
  v8h lo = *(const v8h*)(p);
  v8h hi = *(const v8h*)(p + 16);
  return __builtin_shufflevector(lo, hi, 0,1,2,3,4,5,6,7,8,9,10,11,12,13,14,15);
}

__device__ __forceinline__ v8f wmma16(v16h a, v16h b, v8f c) {
  return __builtin_amdgcn_wmma_f32_16x16x32_f16(false, a, false, b,
                                                (short)0, c, false, false);
}

// ---------------------------------------------------------------------------
__global__ void __launch_bounds__(256)
cvt_f32_to_f16(const float* __restrict__ in, _Float16* __restrict__ out, int n) {
  int i = blockIdx.x * blockDim.x + threadIdx.x;
  int stride = gridDim.x * blockDim.x;
  for (; i < n; i += stride) out[i] = (_Float16)in[i];
}

// ---------------------------------------------------------------------------
// qkv = x @ w_qkv^T + b_qkv. One wave per 64x64 tile: 16 WMMA / k-step.
// Q,K scattered as f16 [b,h,n,d]; V scattered TRANSPOSED as f16 [b,h,d,n].
__global__ void __launch_bounds__(256)
qkv_gemm(const _Float16* __restrict__ A, const _Float16* __restrict__ W,
         const float* __restrict__ bias,
         _Float16* __restrict__ qh, _Float16* __restrict__ kh,
         _Float16* __restrict__ vt) {
  const int K = DIMC;
  const int NT = QKVC / 64;                  // 48 tiles along N
  int wid  = threadIdx.x >> 5;
  int lane = threadIdx.x & 31;
  int half = lane >> 4, r = lane & 15;
  int widx = blockIdx.x * 8 + wid;
  int tm = widx / NT, tn = widx % NT;

  const _Float16* ap[4];
  const _Float16* bp[4];
#pragma unroll
  for (int i = 0; i < 4; ++i) {
    ap[i] = A + (size_t)(tm*64 + i*16 + r) * K + half*8;
    bp[i] = W + (size_t)(tn*64 + i*16 + r) * K + half*8;
  }

  v8f c[4][4];
#pragma unroll
  for (int i = 0; i < 4; ++i)
#pragma unroll
    for (int j = 0; j < 4; ++j) c[i][j] = vzero8f();

  for (int k0 = 0; k0 < K; k0 += 32) {
    __builtin_prefetch(ap[0] + k0 + 64, 0, 3);
    __builtin_prefetch(bp[0] + k0 + 64, 0, 3);
    v16h af[4], bf[4];
#pragma unroll
    for (int i = 0; i < 4; ++i) af[i] = load_a_frag(ap[i] + k0);
#pragma unroll
    for (int j = 0; j < 4; ++j) bf[j] = load_b_frag(bp[j] + k0);
#pragma unroll
    for (int i = 0; i < 4; ++i)
#pragma unroll
      for (int j = 0; j < 4; ++j) c[i][j] = wmma16(af[i], bf[j], c[i][j]);
  }

  // 64-wide tile never straddles a 1024 boundary -> which/head uniform in tile
  int which = tn >> 4;                       // 0:q 1:k 2:v
  int hh    = tn & 15;                       // head
  if (which < 2) {
    _Float16* buf = (which == 0) ? qh : kh;
#pragma unroll
    for (int ni = 0; ni < 4; ++ni) {
      float bv = bias[tn*64 + ni*16 + r];
#pragma unroll
      for (int mi = 0; mi < 4; ++mi)
#pragma unroll
        for (int rr = 0; rr < 8; ++rr) {
          int row = tm*64 + mi*16 + half*8 + rr;
          int bb  = row >> 11;               // / SEQ
          int nn  = row & (SEQ - 1);
          float v = c[mi][ni][rr] + bv;
          buf[(((size_t)bb*HEADS + hh)*SEQ + nn)*HD + ni*16 + r] = (_Float16)v;
        }
    }
  } else {
#pragma unroll
    for (int ni = 0; ni < 4; ++ni) {
      float bv = bias[tn*64 + ni*16 + r];
#pragma unroll
      for (int mi = 0; mi < 4; ++mi)
#pragma unroll
        for (int rr = 0; rr < 8; ++rr) {
          int row = tm*64 + mi*16 + half*8 + rr;
          int bb  = row >> 11;
          int nn  = row & (SEQ - 1);
          float v = c[mi][ni][rr] + bv;
          vt[(((size_t)bb*HEADS + hh)*HD + ni*16 + r)*SEQ + nn] = (_Float16)v;
        }
    }
  }
}

// ---------------------------------------------------------------------------
// Flash attention: one wave per (b, h, 32-row query block) = 2 subtiles of 16.
// Per 32-key chunk: 4 K-frags + 4 V-frags (global, shared by both subtiles),
// 8 S-WMMA + 8 PV-WMMA; masked online softmax in registers; LDS only for the
// P C-layout -> A-layout restage (512 f16 per wave).
__global__ void __launch_bounds__(128)
flash_attn(const _Float16* __restrict__ qh, const _Float16* __restrict__ kh,
           const _Float16* __restrict__ vt, const unsigned char* __restrict__ msk,
           _Float16* __restrict__ ao) {
  __shared__ __align__(16) _Float16 lds[4 * 512];   // P: 16x32 f16 per wave

  int wid  = threadIdx.x >> 5;
  int lane = threadIdx.x & 31;
  int half = lane >> 4, r = lane & 15;
  int w  = blockIdx.x * 4 + wid;
  int qt = w & (SEQ/32 - 1);          // 32-row query block [0,64)
  int h  = (w >> 6) & (HEADS - 1);
  int b  = w >> 10;

  const _Float16* qb  = qh + (size_t)(b*HEADS + h) * SEQ * HD;
  const _Float16* kb  = kh + (size_t)(b*HEADS + h) * SEQ * HD;
  const _Float16* vtb = vt + (size_t)(b*HEADS + h) * HD * SEQ;  // [d][n]
  const unsigned char* mb = msk + (size_t)b * SEQ * SEQ;
  _Float16* P = lds + wid * 512;

  int qrow0 = qt * 32;

  // Q fragments: 2 subtiles x 2 K-steps (resident for the whole loop)
  v16h q[2][2];
#pragma unroll
  for (int st = 0; st < 2; ++st)
#pragma unroll
    for (int ks = 0; ks < 2; ++ks)
      q[st][ks] = load_a_frag(qb + (size_t)(qrow0 + st*16 + r)*HD + ks*32 + half*8);

  v8f o[2][4];
  float mrow[2][8], lrow[2][8];
#pragma unroll
  for (int st = 0; st < 2; ++st) {
#pragma unroll
    for (int dt = 0; dt < 4; ++dt) o[st][dt] = vzero8f();
#pragma unroll
    for (int i = 0; i < 8; ++i) { mrow[st][i] = NEG_INF; lrow[st][i] = 0.f; }
  }
  const float scale = 0.125f;  // 1/sqrt(64)

  for (int n0 = 0; n0 < SEQ; n0 += 32) {
    __builtin_prefetch(kb + (size_t)(n0 + 32 + r)*HD, 0, 3);
    // K^T B-fragments: lane = key column, d contiguous per half
    v16h kf[2][2];
#pragma unroll
    for (int f = 0; f < 2; ++f)
#pragma unroll
      for (int ks = 0; ks < 2; ++ks)
        kf[f][ks] = load_b_frag(kb + (size_t)(n0 + f*16 + r)*HD + ks*32 + half*8);
    // V B-fragments from transposed layout: lane = d column, keys contiguous
    v16h vf[4];
#pragma unroll
    for (int dt = 0; dt < 4; ++dt)
      vf[dt] = load_b_frag(vtb + (size_t)(dt*16 + r)*SEQ + n0 + half*8);

#pragma unroll
    for (int st = 0; st < 2; ++st) {
      // S = Q K^T (16x32 in two C fragments)
      v8f s0 = vzero8f(), s1 = vzero8f();
      s0 = wmma16(q[st][0], kf[0][0], s0); s0 = wmma16(q[st][1], kf[0][1], s0);
      s1 = wmma16(q[st][0], kf[1][0], s1); s1 = wmma16(q[st][1], kf[1][1], s1);

      // mask + online softmax (C layout: vgpr rr -> row rr+half*8)
#pragma unroll
      for (int rr = 0; rr < 8; ++rr) {
        int qrow = qrow0 + st*16 + rr + half*8;
        const unsigned char* mp = mb + (size_t)qrow * SEQ + n0;
        float v0 = s0[rr] * scale;
        float v1 = s1[rr] * scale;
        if (mp[r])      v0 = NEG_INF;
        if (mp[16 + r]) v1 = NEG_INF;
        float vmax = fmaxf(v0, v1);
#pragma unroll
        for (int sh = 1; sh < 16; sh <<= 1)
          vmax = fmaxf(vmax, __shfl_xor(vmax, sh, 32));
        float mnew  = fmaxf(mrow[st][rr], vmax);
        float alpha = __expf(mrow[st][rr] - mnew);
        float p0 = __expf(v0 - mnew);
        float p1 = __expf(v1 - mnew);
        float rs = p0 + p1;
#pragma unroll
        for (int sh = 1; sh < 16; sh <<= 1)
          rs += __shfl_xor(rs, sh, 32);
        lrow[st][rr] = lrow[st][rr] * alpha + rs;
        mrow[st][rr] = mnew;
        o[st][0][rr] *= alpha; o[st][1][rr] *= alpha;
        o[st][2][rr] *= alpha; o[st][3][rr] *= alpha;
        int prow = rr + half*8;
        P[prow*32 + r]      = (_Float16)p0;
        P[prow*32 + 16 + r] = (_Float16)p1;
      }

      // O += P(16x32) @ V(32x64); LDS ops are in-order within a wave
      v16h pa = load_a_frag(P + r*32 + half*8);
#pragma unroll
      for (int dt = 0; dt < 4; ++dt) o[st][dt] = wmma16(pa, vf[dt], o[st][dt]);
    }
  }

  // normalize, write attn_out [b, n, h*64 + d] f16
#pragma unroll
  for (int st = 0; st < 2; ++st)
#pragma unroll
    for (int dt = 0; dt < 4; ++dt)
#pragma unroll
      for (int rr = 0; rr < 8; ++rr) {
        int nrow = qrow0 + st*16 + rr + half*8;
        int col  = h*HD + dt*16 + r;
        float v = o[st][dt][rr] / lrow[st][rr];
        ao[((size_t)b*SEQ + nrow)*DIMC + col] = (_Float16)v;
      }
}

// ---------------------------------------------------------------------------
// out = attn_out @ w_proj^T + b_proj (f32 out). 64x64 tile per wave.
__global__ void __launch_bounds__(256)
proj_gemm(const _Float16* __restrict__ A, const _Float16* __restrict__ W,
          const float* __restrict__ bias, float* __restrict__ out) {
  const int K = DIMC;
  const int NT = DIMC / 64;                  // 16 tiles along N
  int wid  = threadIdx.x >> 5;
  int lane = threadIdx.x & 31;
  int half = lane >> 4, r = lane & 15;
  int widx = blockIdx.x * 8 + wid;
  int tm = widx / NT, tn = widx % NT;

  const _Float16* ap[4];
  const _Float16* bp[4];
#pragma unroll
  for (int i = 0; i < 4; ++i) {
    ap[i] = A + (size_t)(tm*64 + i*16 + r) * K + half*8;
    bp[i] = W + (size_t)(tn*64 + i*16 + r) * K + half*8;
  }

  v8f c[4][4];
#pragma unroll
  for (int i = 0; i < 4; ++i)
#pragma unroll
    for (int j = 0; j < 4; ++j) c[i][j] = vzero8f();

  for (int k0 = 0; k0 < K; k0 += 32) {
    __builtin_prefetch(ap[0] + k0 + 64, 0, 3);
    __builtin_prefetch(bp[0] + k0 + 64, 0, 3);
    v16h af[4], bf[4];
#pragma unroll
    for (int i = 0; i < 4; ++i) af[i] = load_a_frag(ap[i] + k0);
#pragma unroll
    for (int j = 0; j < 4; ++j) bf[j] = load_b_frag(bp[j] + k0);
#pragma unroll
    for (int i = 0; i < 4; ++i)
#pragma unroll
      for (int j = 0; j < 4; ++j) c[i][j] = wmma16(af[i], bf[j], c[i][j]);
  }

#pragma unroll
  for (int ni = 0; ni < 4; ++ni) {
    int col = tn*64 + ni*16 + r;
    float bv = bias[col];
#pragma unroll
    for (int mi = 0; mi < 4; ++mi)
#pragma unroll
      for (int rr = 0; rr < 8; ++rr) {
        int row = tm*64 + mi*16 + half*8 + rr;
        out[(size_t)row*DIMC + col] = c[mi][ni][rr] + bv;
      }
  }
}

// ---------------------------------------------------------------------------
extern "C" void kernel_launch(void* const* d_in, const int* in_sizes, int n_in,
                              void* d_out, int out_size, void* d_ws, size_t ws_size,
                              hipStream_t stream) {
  const float* x      = (const float*)d_in[0];
  const float* w_qkv  = (const float*)d_in[1];
  const float* b_qkv  = (const float*)d_in[2];
  const float* w_proj = (const float*)d_in[3];
  const float* b_proj = (const float*)d_in[4];
  const unsigned char* mask = (const unsigned char*)d_in[5];
  float* out = (float*)d_out;

  const size_t M = (size_t)BATCH * SEQ;          // 8192
  char* ws = (char*)d_ws;
  size_t off = 0;
  auto carve = [&](size_t elems) -> _Float16* {
    _Float16* p = (_Float16*)(ws + off);
    off += (elems * sizeof(_Float16) + 255) & ~(size_t)255;
    return p;
  };
  _Float16* xh     = carve(M * DIMC);            // 16 MB
  _Float16* wqkvh  = carve((size_t)QKVC * DIMC); //  6 MB
  _Float16* wprojh = carve((size_t)DIMC * DIMC); //  2 MB
  _Float16* qh     = carve(M * DIMC);            // 16 MB
  _Float16* kh     = carve(M * DIMC);            // 16 MB
  _Float16* vth    = carve(M * DIMC);            // 16 MB (transposed [b,h,d,n])
  _Float16* aoh    = carve(M * DIMC);            // 16 MB
  (void)ws_size; (void)n_in; (void)in_sizes; (void)out_size;

  // 1) f32 -> f16 operand conversion (bandwidth-bound, tiny vs GEMMs)
  cvt_f32_to_f16<<<2048, 256, 0, stream>>>(x,      xh,     (int)(M * DIMC));
  cvt_f32_to_f16<<<2048, 256, 0, stream>>>(w_qkv,  wqkvh,  QKVC * DIMC);
  cvt_f32_to_f16<<<2048, 256, 0, stream>>>(w_proj, wprojh, DIMC * DIMC);

  // 2) QKV projection: waves = (8192/64)*(3072/64) = 6144 -> 768 blocks
  qkv_gemm<<<768, 256, 0, stream>>>(xh, wqkvh, b_qkv, qh, kh, vth);

  // 3) Flash attention: waves = 4*16*(2048/32) = 4096 -> 1024 blocks
  flash_attn<<<1024, 128, 0, stream>>>(qh, kh, vth, mask, aoh);

  // 4) Output projection: waves = (8192/64)*(1024/64) = 2048 -> 256 blocks
  proj_gemm<<<256, 256, 0, stream>>>(aoh, wprojh, b_proj, out);
}
